// SimpleNet_40638980555308
// MI455X (gfx1250) — matrix operand order repaired and checked by reading, not compiled, not run on hardware
//
#include <hip/hip_runtime.h>

#define F_IN 512
#define HID  16
#define NCLS 4

typedef __attribute__((ext_vector_type(16))) __bf16 v16bf;
typedef __attribute__((ext_vector_type(16))) float  v16f;
typedef __attribute__((ext_vector_type(8)))  float  v8f;

__device__ __forceinline__ unsigned short f32_to_bf16(float f) {
  unsigned u = __float_as_uint(f);
  unsigned r = u + 0x7FFFu + ((u >> 16) & 1u);  // round-to-nearest-even
  return (unsigned short)(r >> 16);
}

// ---------------- init: deg = 1 (self loop), zero accumulators ----------------
__global__ void init_kernel(float* __restrict__ deg, float* __restrict__ out1,
                            float* __restrict__ out2, int N) {
  int i = blockIdx.x * blockDim.x + threadIdx.x;
  if (i < N) deg[i] = 1.0f;
  if (i < N * HID) out1[i] = 0.0f;
  if (i < N * NCLS) out2[i] = 0.0f;
}

// ---------------- degree: deg[dst] += 1 per edge ----------------
__global__ void degree_kernel(const int* __restrict__ ei, float* __restrict__ deg, int E) {
  int i = blockIdx.x * blockDim.x + threadIdx.x;
  if (i < E) atomicAdd(&deg[ei[E + i]], 1.0f);
}

// ---------------- dinv = rsqrt(deg) in place ----------------
__global__ void rsqrt_kernel(float* __restrict__ deg, int N) {
  int i = blockIdx.x * blockDim.x + threadIdx.x;
  if (i < N) deg[i] = rsqrtf(deg[i]);
}

// ---------------- pack W1 (512x16 f32) into bf16 WMMA B fragments ----------------
// Layout: Bpack[chunk][lane][8 dwords]; chunk c covers K = 32c..32c+31.
// Lane n (<16): K = 32c+0..15 (2 bf16/dword); lane n+16: K = 32c+16..31.
__global__ void pack_w1_kernel(const float* __restrict__ W1, unsigned* __restrict__ Bpack) {
  int t = threadIdx.x;                 // 512 threads: 16 chunks x 32 lanes
  int c = t >> 5, lane = t & 31;
  int n = lane & 15;
  int klo = (lane >> 4) * 16;
  unsigned* dst = Bpack + (size_t)(c * 32 + lane) * 8;
  #pragma unroll
  for (int v = 0; v < 8; ++v) {
    int k = c * 32 + klo + 2 * v;
    unsigned lo = f32_to_bf16(W1[(size_t)k * HID + n]);
    unsigned hi = f32_to_bf16(W1[(size_t)(k + 1) * HID + n]);
    dst[v] = lo | (hi << 16);
  }
}

// ---------------- GEMM1: h1 = x @ W1 via v_wmma_f32_16x16x32_bf16 ----------------
// One wave per 16-row tile; K = 512 -> 16 WMMA ops.
__global__ void __launch_bounds__(256)
gemm1_wmma_kernel(const float* __restrict__ x, const unsigned* __restrict__ Bpack,
                  float* __restrict__ h1, int nTiles, int N) {
  int wave = (blockIdx.x * blockDim.x + threadIdx.x) >> 5;
  int lane = threadIdx.x & 31;
  if (wave >= nTiles) return;                 // wave-uniform: EXEC stays all-ones for WMMA

  const int rowBase = wave * 16;
  const int hi = lane >> 4;                   // lane group selects K sub-range
  const int n  = lane & 15;                   // A row within tile / B column
  int arow = rowBase + n; if (arow >= N) arow = N - 1;   // clamp (tail safety)
  const float* xrow = x + (size_t)arow * F_IN;

  v8f acc = {};
  #pragma unroll
  for (int c = 0; c < 16; ++c) {
    const int kc = c * 32;
    // A fragment: lane holds K = kc + hi*16 + {0..15}
    const float4* p0 = (const float4*)(xrow + kc + hi * 16);
    const float4* p1 = (const float4*)(xrow + kc + hi * 16 + 8);
    union { float4 f4[4]; v16f v; } af;
    af.f4[0] = p0[0]; af.f4[1] = p0[1];
    af.f4[2] = p1[0]; af.f4[3] = p1[1];
    // native hardware f32 -> bf16 truncation (RNE); lets backend pick packed cvt
    v16bf a = __builtin_convertvector(af.v, v16bf);
    // B fragment: 32 contiguous bytes per lane (L2/L0-resident after first tile)
    union { v16bf v; uint4 q[2]; } B;
    const uint4* bp = (const uint4*)(Bpack + (size_t)(c * 32 + lane) * 8);
    B.q[0] = bp[0]; B.q[1] = bp[1];
    acc = __builtin_amdgcn_wmma_f32_16x16x32_bf16(false, a, false, B.v,
                                                  (short)0, acc, false, false);
  }
  // C layout: VGPR r holds row (r + 8*hi), column n
  #pragma unroll
  for (int r = 0; r < 8; ++r) {
    int m = rowBase + hi * 8 + r;
    if (m < N) h1[(size_t)m * HID + n] = acc[r];
  }
}

// ---------------- aggregation 1: 16 lanes per (edge|self-loop) ----------------
__global__ void agg1_kernel(const int* __restrict__ ei, const float* __restrict__ dinv,
                            const float* __restrict__ h1, float* __restrict__ out1,
                            int E, int N) {
  int idx = blockIdx.x * blockDim.x + threadIdx.x;
  int g = idx >> 4, j = idx & 15;
  if (g >= E + N) return;
  int s, d;
  if (g < E) { s = ei[g]; d = ei[E + g]; } else { s = d = g - E; }
  float nrm = dinv[s] * dinv[d];
  atomicAdd(&out1[(size_t)d * HID + j], h1[(size_t)s * HID + j] * nrm);
}

// ---------------- layer 2: h2 = relu(out1 + b1) @ W2 ----------------
__global__ void layer2_kernel(const float* __restrict__ out1, const float* __restrict__ b1,
                              const float* __restrict__ W2, float* __restrict__ h2, int N) {
  int i = blockIdx.x * blockDim.x + threadIdx.x;
  if (i >= N) return;
  float h[HID];
  const float* row = out1 + (size_t)i * HID;
  #pragma unroll
  for (int j = 0; j < HID; ++j) h[j] = fmaxf(row[j] + b1[j], 0.0f);
  float a0 = 0.f, a1 = 0.f, a2 = 0.f, a3 = 0.f;
  #pragma unroll
  for (int j = 0; j < HID; ++j) {
    const float* w = W2 + j * NCLS;
    a0 = fmaf(h[j], w[0], a0); a1 = fmaf(h[j], w[1], a1);
    a2 = fmaf(h[j], w[2], a2); a3 = fmaf(h[j], w[3], a3);
  }
  float4 r = make_float4(a0, a1, a2, a3);
  *(float4*)(h2 + (size_t)i * NCLS) = r;
}

// ---------------- aggregation 2: 4 lanes per (edge|self-loop) ----------------
__global__ void agg2_kernel(const int* __restrict__ ei, const float* __restrict__ dinv,
                            const float* __restrict__ h2, float* __restrict__ out2,
                            int E, int N) {
  int idx = blockIdx.x * blockDim.x + threadIdx.x;
  int g = idx >> 2, j = idx & 3;
  if (g >= E + N) return;
  int s, d;
  if (g < E) { s = ei[g]; d = ei[E + g]; } else { s = d = g - E; }
  float nrm = dinv[s] * dinv[d];
  atomicAdd(&out2[(size_t)d * NCLS + j], h2[(size_t)s * NCLS + j] * nrm);
}

// ---------------- bias + log_softmax ----------------
__global__ void logsoftmax_kernel(const float* __restrict__ out2, const float* __restrict__ b2,
                                  float* __restrict__ out, int N) {
  int i = blockIdx.x * blockDim.x + threadIdx.x;
  if (i >= N) return;
  const float* r = out2 + (size_t)i * NCLS;
  float v0 = r[0] + b2[0], v1 = r[1] + b2[1], v2 = r[2] + b2[2], v3 = r[3] + b2[3];
  float m = fmaxf(fmaxf(v0, v1), fmaxf(v2, v3));
  float se = expf(v0 - m) + expf(v1 - m) + expf(v2 - m) + expf(v3 - m);
  float ls = m + logf(se);
  float* o = out + (size_t)i * NCLS;
  o[0] = v0 - ls; o[1] = v1 - ls; o[2] = v2 - ls; o[3] = v3 - ls;
}

extern "C" void kernel_launch(void* const* d_in, const int* in_sizes, int n_in,
                              void* d_out, int out_size, void* d_ws, size_t ws_size,
                              hipStream_t stream) {
  const float* x  = (const float*)d_in[0];
  const int*   ei = (const int*)d_in[1];
  const float* W1 = (const float*)d_in[2];
  const float* b1 = (const float*)d_in[3];
  const float* W2 = (const float*)d_in[4];
  const float* b2 = (const float*)d_in[5];
  float* out = (float*)d_out;

  const int N = in_sizes[0] / F_IN;   // 100000
  const int E = in_sizes[1] / 2;      // 3200000

  // workspace layout (floats)
  float* ws = (float*)d_ws;
  float*    deg   = ws;                                   // N  (becomes dinv)
  float*    h1    = ws + (size_t)N;                       // N*16
  float*    out1  = h1 + (size_t)N * HID;                 // N*16
  float*    h2    = out1 + (size_t)N * HID;               // N*4
  float*    out2  = h2 + (size_t)N * NCLS;                // N*4
  unsigned* Bpack = (unsigned*)(out2 + (size_t)N * NCLS); // 16*32*8 dwords

  const int TB = 256;
  init_kernel<<<(N * HID + TB - 1) / TB, TB, 0, stream>>>(deg, out1, out2, N);
  degree_kernel<<<(E + TB - 1) / TB, TB, 0, stream>>>(ei, deg, E);
  rsqrt_kernel<<<(N + TB - 1) / TB, TB, 0, stream>>>(deg, N);
  pack_w1_kernel<<<1, 512, 0, stream>>>(W1, Bpack);

  const int nTiles = (N + 15) / 16;
  gemm1_wmma_kernel<<<(nTiles * 32 + TB - 1) / TB, TB, 0, stream>>>(x, Bpack, h1, nTiles, N);

  const long long w1t = (long long)(E + N) * HID;
  agg1_kernel<<<(int)((w1t + TB - 1) / TB), TB, 0, stream>>>(ei, deg, h1, out1, E, N);
  layer2_kernel<<<(N + TB - 1) / TB, TB, 0, stream>>>(out1, b1, W2, h2, N);
  const long long w2t = (long long)(E + N) * NCLS;
  agg2_kernel<<<(int)((w2t + TB - 1) / TB), TB, 0, stream>>>(ei, deg, h2, out2, E, N);
  logsoftmax_kernel<<<(N + TB - 1) / TB, TB, 0, stream>>>(out2, b2, out, N);
}
